// Colar_static_46265387713092
// MI455X (gfx1250) — matrix operand order repaired and checked by reading, not compiled, not run on hardware
//
#include <hip/hip_runtime.h>

typedef __bf16 bf16_t;
typedef bf16_t v16bf __attribute__((ext_vector_type(16)));
typedef bf16_t v8bf  __attribute__((ext_vector_type(8)));
typedef float  v8f   __attribute__((ext_vector_type(8)));

#define B_   4096
#define T_   8
#define CHIN 2048
#define C_   1024
#define N_   512
#define K_   5
#define KN_  (K_ * N_)   // 2560
#define EPS_ 1e-8f

// ---------------- conversion kernels ----------------

__global__ __launch_bounds__(256) void conv_xlast(const float* __restrict__ x,
                                                  bf16_t* __restrict__ xb) {
    size_t i  = (size_t)blockIdx.x * 256 + threadIdx.x;   // 0 .. B_*CHIN-1
    size_t b  = i / CHIN;
    size_t ch = i % CHIN;
    xb[i] = (bf16_t)x[(b * T_ + (T_ - 1)) * CHIN + ch];
}

__global__ __launch_bounds__(256) void conv_f32_bf16(const float* __restrict__ s,
                                                     bf16_t* __restrict__ d, int n) {
    int i = blockIdx.x * 256 + threadIdx.x;
    if (i < n) d[i] = (bf16_t)s[i];
}

// static (K, CHIN, N) f32 -> staticT (K, N, CHIN) bf16 via LDS tile transpose
__global__ __launch_bounds__(256) void transpose_static(const float* __restrict__ s,
                                                        bf16_t* __restrict__ d) {
    __shared__ float tile[32][33];
    const int kk = blockIdx.z;
    const int i0 = blockIdx.y * 32;   // CHIN tile base
    const int n0 = blockIdx.x * 32;   // N tile base
    const int tx = threadIdx.x & 31;
    const int ty = threadIdx.x >> 5;  // 0..7
    const float* src = s + (size_t)kk * CHIN * N_;
    #pragma unroll
    for (int r = 0; r < 4; ++r)
        tile[ty + r * 8][tx] = src[(size_t)(i0 + ty + r * 8) * N_ + n0 + tx];
    __syncthreads();
    bf16_t* dst = d + (size_t)kk * N_ * CHIN;
    #pragma unroll
    for (int r = 0; r < 4; ++r)
        dst[(size_t)(n0 + ty + r * 8) * CHIN + i0 + tx] = (bf16_t)tile[tx][ty + r * 8];
}

// ---------------- generic bf16 WMMA GEMM ----------------
// D(M,N) = A(M,K) @ B(K,N) [+ bias], A row-major (lda), B given as Bt (N,K) row-major (ldb).
// Bt gets +z*bt_z, D gets +z*d_z (element offsets).
// BIAS_MODE: 0 none, 1 per-column, 2 per-row.  OUT_BF16: store bf16 instead of f32.
template <int BIAS_MODE, bool OUT_BF16>
__global__ __launch_bounds__(256)
void gemm_bf16(const bf16_t* __restrict__ A, int lda,
               const bf16_t* __restrict__ Bt, int ldb, long bt_z,
               void* __restrict__ Dv, int ldd, long d_z,
               const float* __restrict__ bias,
               int M, int N, int K) {
    const int lane = threadIdx.x & 31;
    const int wave = threadIdx.x >> 5;       // 0..7
    const int wm   = wave & 1;               // 2 waves in M
    const int wn   = wave >> 1;              // 4 waves in N
    const int m0   = blockIdx.y * 64  + wm * 32;
    const int n0   = blockIdx.x * 256 + wn * 64;
    const int z    = blockIdx.z;
    const int l15  = lane & 15;
    const int lhi  = lane >> 4;

    const bf16_t* Bz = Bt + (size_t)z * bt_z;

    v8f acc[2][4];
    #pragma unroll
    for (int m = 0; m < 2; ++m)
        #pragma unroll
        for (int n = 0; n < 4; ++n)
            #pragma unroll
            for (int r = 0; r < 8; ++r) acc[m][n][r] = 0.0f;

    for (int kt = 0; kt < K; kt += 32) {
        // A fragments: lane holds row m0+m*16+l15, k = kbase+{0..7} and kbase+{16..23}
        v16bf af[2];
        #pragma unroll
        for (int m = 0; m < 2; ++m) {
            const bf16_t* pa = A + (size_t)(m0 + m * 16 + l15) * lda + kt + lhi * 8;
            union { v16bf v; v8bf h[2]; } u;
            u.h[0] = *(const v8bf*)(pa);
            u.h[1] = *(const v8bf*)(pa + 16);
            af[m] = u.v;
        }
        // B fragments: lane holds column n0+n*16+l15, k = kbase+{0..7} and kbase+{16..23}
        v16bf bfr[4];
        #pragma unroll
        for (int n = 0; n < 4; ++n) {
            const bf16_t* pb = Bz + (size_t)(n0 + n * 16 + l15) * ldb + kt + lhi * 8;
            union { v16bf v; v8bf h[2]; } ub;
            ub.h[0] = *(const v8bf*)(pb);
            ub.h[1] = *(const v8bf*)(pb + 16);
            bfr[n] = ub.v;
        }
        #pragma unroll
        for (int m = 0; m < 2; ++m)
            #pragma unroll
            for (int n = 0; n < 4; ++n)
                acc[m][n] = __builtin_amdgcn_wmma_f32_16x16x32_bf16(
                    false, af[m], false, bfr[n], (short)0, acc[m][n], false, false);
    }

    // per-row bias (rows are consecutive for r=0..7 at base m0+m*16+lhi*8)
    float brow[2][8];
    if (BIAS_MODE == 2) {
        #pragma unroll
        for (int m = 0; m < 2; ++m)
            #pragma unroll
            for (int r = 0; r < 8; ++r)
                brow[m][r] = bias[m0 + m * 16 + lhi * 8 + r];
    }

    // epilogue: D layout — VGPR r: row = r + 8*lhi; col = l15
    float*  Df = (float*)Dv;
    bf16_t* Dh = (bf16_t*)Dv;
    #pragma unroll
    for (int m = 0; m < 2; ++m) {
        #pragma unroll
        for (int n = 0; n < 4; ++n) {
            const int col = n0 + n * 16 + l15;
            const float bcol = (BIAS_MODE == 1) ? bias[col] : 0.0f;
            #pragma unroll
            for (int r = 0; r < 8; ++r) {
                const int row = m0 + m * 16 + r + lhi * 8;
                float v = acc[m][n][r];
                if (BIAS_MODE == 1) v += bcol;
                if (BIAS_MODE == 2) v += brow[m][r];
                const size_t idx = (size_t)row * ldd + col + (size_t)z * d_z;
                if (OUT_BF16) Dh[idx] = (bf16_t)v;
                else          Df[idx] = v;
            }
        }
    }
}

// ---------------- normalization / softmax / epilogue kernels ----------------

// kn = k / max(||k||_2, eps), row-wise; output bf16
__global__ __launch_bounds__(256) void kn_kernel(const float* __restrict__ kf,
                                                 bf16_t* __restrict__ knb) {
    __shared__ float red[256];
    const int b = blockIdx.x, t = threadIdx.x;
    float v[4], s = 0.f;
    #pragma unroll
    for (int i = 0; i < 4; ++i) {
        v[i] = kf[(size_t)b * C_ + t + i * 256];
        s += v[i] * v[i];
    }
    red[t] = s; __syncthreads();
    for (int w = 128; w > 0; w >>= 1) { if (t < w) red[t] += red[t + w]; __syncthreads(); }
    const float inv = 1.0f / fmaxf(sqrtf(red[0]), EPS_);
    #pragma unroll
    for (int i = 0; i < 4; ++i)
        knb[(size_t)b * C_ + t + i * 256] = (bf16_t)(v[i] * inv);
}

// Ekn columns: normalize Ekf[kk][:,n] over c, write transposed Eknt[kk][n][c] bf16
__global__ __launch_bounds__(256) void ekn_kernel(const float* __restrict__ Ekf,
                                                  bf16_t* __restrict__ Eknt) {
    __shared__ float red[256];
    const int kn = blockIdx.x;           // kk*N_ + n
    const int kk = kn / N_, n = kn % N_;
    const int t = threadIdx.x;
    const float* src = Ekf + (size_t)kk * C_ * N_ + n;
    float v[4], s = 0.f;
    #pragma unroll
    for (int i = 0; i < 4; ++i) {
        v[i] = src[(size_t)(t + i * 256) * N_];
        s += v[i] * v[i];
    }
    red[t] = s; __syncthreads();
    for (int w = 128; w > 0; w >>= 1) { if (t < w) red[t] += red[t + w]; __syncthreads(); }
    const float inv = 1.0f / fmaxf(sqrtf(red[0]), EPS_);
    bf16_t* dst = Eknt + (size_t)kk * N_ * C_ + (size_t)n * C_;
    #pragma unroll
    for (int i = 0; i < 4; ++i)
        dst[t + i * 256] = (bf16_t)(v[i] * inv);
}

// g[k,n] = sum_c Ww[c] * Ev[k,c,n]   (EvT stored (C, K*N))
__global__ __launch_bounds__(256) void g_kernel(const bf16_t* __restrict__ EvT,
                                                const float* __restrict__ Ww,
                                                float* __restrict__ g) {
    __shared__ float red[256];
    const int idx = blockIdx.x;          // 0..KN_-1
    const int t = threadIdx.x;
    float s = 0.f;
    #pragma unroll
    for (int i = 0; i < 4; ++i) {
        const int c = t + i * 256;
        s += Ww[c] * (float)EvT[(size_t)c * KN_ + idx];
    }
    red[t] = s; __syncthreads();
    for (int w = 128; w > 0; w >>= 1) { if (t < w) red[t] += red[t + w]; __syncthreads(); }
    if (t == 0) g[idx] = red[0];
}

// per (b,kk) row: w = softmax(sim); fw = sigmoid(w.g + bw); wprime = w*fw (bf16)
__global__ __launch_bounds__(256) void softmax_wprime(const float* __restrict__ simf,
                                                      const float* __restrict__ g,
                                                      const float* __restrict__ bw,
                                                      bf16_t* __restrict__ wpr) {
    __shared__ float red[256];
    const int row = blockIdx.x;          // b*K_ + kk
    const int kk = row % K_;
    const int t = threadIdx.x;
    const float* s = simf + (size_t)row * N_;
    const float a0 = s[t], a1 = s[t + 256];

    red[t] = fmaxf(a0, a1); __syncthreads();
    for (int w = 128; w > 0; w >>= 1) { if (t < w) red[t] = fmaxf(red[t], red[t + w]); __syncthreads(); }
    const float mx = red[0]; __syncthreads();

    const float e0 = __expf(a0 - mx), e1 = __expf(a1 - mx);
    red[t] = e0 + e1; __syncthreads();
    for (int w = 128; w > 0; w >>= 1) { if (t < w) red[t] += red[t + w]; __syncthreads(); }
    const float inv = 1.0f / red[0]; __syncthreads();

    const float w0 = e0 * inv, w1 = e1 * inv;
    const float* gk = g + kk * N_;
    red[t] = w0 * gk[t] + w1 * gk[t + 256]; __syncthreads();
    for (int w = 128; w > 0; w >>= 1) { if (t < w) red[t] += red[t + w]; __syncthreads(); }
    const float fw = 1.0f / (1.0f + __expf(-(red[0] + bw[0])));

    bf16_t* o = wpr + (size_t)row * N_;
    o[t]       = (bf16_t)(w0 * fw);
    o[t + 256] = (bf16_t)(w1 * fw);
}

// out[b,kk] = bout[kk] + Wout[kk,0:C].relu(v[b]) + Wout[kk,C:2C].relu(fE[b])
__global__ __launch_bounds__(256) void final_kernel(const float* __restrict__ vf,
                                                    const float* __restrict__ fEf,
                                                    const float* __restrict__ Wout,
                                                    const float* __restrict__ bout,
                                                    float* __restrict__ out) {
    __shared__ float red[K_][256];
    const int b = blockIdx.x, t = threadIdx.x;
    float acc[K_];
    #pragma unroll
    for (int k = 0; k < K_; ++k) acc[k] = 0.f;
    for (int c = t; c < C_; c += 256) {
        const float rv = fmaxf(vf[(size_t)b * C_ + c], 0.f);
        const float rf = fmaxf(fEf[(size_t)b * C_ + c], 0.f);
        #pragma unroll
        for (int k = 0; k < K_; ++k)
            acc[k] += Wout[k * 2 * C_ + c] * rv + Wout[k * 2 * C_ + C_ + c] * rf;
    }
    #pragma unroll
    for (int k = 0; k < K_; ++k) red[k][t] = acc[k];
    __syncthreads();
    for (int w = 128; w > 0; w >>= 1) {
        if (t < w) {
            #pragma unroll
            for (int k = 0; k < K_; ++k) red[k][t] += red[k][t + w];
        }
        __syncthreads();
    }
    if (t < K_) out[(size_t)b * K_ + t] = red[t][0] + bout[t];
}

// ---------------- host launcher ----------------

extern "C" void kernel_launch(void* const* d_in, const int* in_sizes, int n_in,
                              void* d_out, int out_size, void* d_ws, size_t ws_size,
                              hipStream_t stream) {
    const float* x    = (const float*)d_in[0];
    const float* stat = (const float*)d_in[1];
    const float* Wk   = (const float*)d_in[2];
    const float* bk   = (const float*)d_in[3];
    const float* Wv   = (const float*)d_in[4];
    const float* bv   = (const float*)d_in[5];
    const float* WEk  = (const float*)d_in[6];
    const float* bEk  = (const float*)d_in[7];
    const float* WEv  = (const float*)d_in[8];
    const float* bEv  = (const float*)d_in[9];
    const float* Ww   = (const float*)d_in[10];
    const float* bw   = (const float*)d_in[11];
    const float* Wout = (const float*)d_in[12];
    const float* bout = (const float*)d_in[13];
    float* out = (float*)d_out;

    char* ws = (char*)d_ws;
    size_t off = 0;
    auto alloc = [&](size_t bytes) -> char* {
        char* p = ws + off;
        off += (bytes + 255) & ~(size_t)255;
        return p;
    };

    bf16_t* xb      = (bf16_t*)alloc((size_t)B_ * CHIN * 2);
    bf16_t* Wkb     = (bf16_t*)alloc((size_t)C_ * CHIN * 2);
    bf16_t* Wvb     = (bf16_t*)alloc((size_t)C_ * CHIN * 2);
    bf16_t* WEkb    = (bf16_t*)alloc((size_t)C_ * CHIN * 2);
    bf16_t* WEvb    = (bf16_t*)alloc((size_t)C_ * CHIN * 2);
    bf16_t* staticT = (bf16_t*)alloc((size_t)K_ * N_ * CHIN * 2);
    float*  kf      = (float*) alloc((size_t)B_ * C_ * 4);
    float*  vf      = (float*) alloc((size_t)B_ * C_ * 4);
    float*  Ekf     = (float*) alloc((size_t)K_ * C_ * N_ * 4);
    bf16_t* EvT     = (bf16_t*)alloc((size_t)C_ * KN_ * 2);
    bf16_t* knb     = (bf16_t*)alloc((size_t)B_ * C_ * 2);
    bf16_t* Eknt    = (bf16_t*)alloc((size_t)K_ * N_ * C_ * 2);
    float*  simf    = (float*) alloc((size_t)B_ * KN_ * 4);
    float*  gbuf    = (float*) alloc((size_t)KN_ * 4);
    bf16_t* wprimeb = (bf16_t*)alloc((size_t)B_ * KN_ * 2);
    float*  fEf     = (float*) alloc((size_t)B_ * C_ * 4);

    // 1) conversions
    conv_xlast<<<(B_ * (size_t)CHIN) / 256, 256, 0, stream>>>(x, xb);
    const int wsz = C_ * CHIN;
    conv_f32_bf16<<<wsz / 256, 256, 0, stream>>>(Wk,  Wkb,  wsz);
    conv_f32_bf16<<<wsz / 256, 256, 0, stream>>>(Wv,  Wvb,  wsz);
    conv_f32_bf16<<<wsz / 256, 256, 0, stream>>>(WEk, WEkb, wsz);
    conv_f32_bf16<<<wsz / 256, 256, 0, stream>>>(WEv, WEvb, wsz);
    transpose_static<<<dim3(N_ / 32, CHIN / 32, K_), 256, 0, stream>>>(stat, staticT);

    // 2) k = x W_k^T + b_k ; v = x W_v^T + b_v   (f32 out, per-column bias)
    dim3 gkv(C_ / 256, B_ / 64, 1);
    gemm_bf16<1, false><<<gkv, 256, 0, stream>>>(xb, CHIN, Wkb, CHIN, 0,
                                                 kf, C_, 0, bk, B_, C_, CHIN);
    gemm_bf16<1, false><<<gkv, 256, 0, stream>>>(xb, CHIN, Wvb, CHIN, 0,
                                                 vf, C_, 0, bv, B_, C_, CHIN);

    // 3) Ek = WEk @ static[k] + bEk (f32) ; Ev likewise, written transposed bf16 to EvT
    dim3 gE(N_ / 256, C_ / 64, K_);
    gemm_bf16<2, false><<<gE, 256, 0, stream>>>(WEkb, CHIN, staticT, CHIN, (long)N_ * CHIN,
                                                Ekf, N_, (long)C_ * N_, bEk, C_, N_, CHIN);
    gemm_bf16<2, true><<<gE, 256, 0, stream>>>(WEvb, CHIN, staticT, CHIN, (long)N_ * CHIN,
                                               EvT, KN_, (long)N_, bEv, C_, N_, CHIN);

    // 4) normalizations + gating precompute
    kn_kernel<<<B_, 256, 0, stream>>>(kf, knb);
    ekn_kernel<<<K_ * N_, 256, 0, stream>>>(Ekf, Eknt);
    g_kernel<<<KN_, 256, 0, stream>>>(EvT, Ww, gbuf);

    // 5) sim[b,kk,n] = kn . Ekn
    dim3 gs(N_ / 256, B_ / 64, K_);
    gemm_bf16<0, false><<<gs, 256, 0, stream>>>(knb, C_, Eknt, C_, (long)N_ * C_,
                                                simf, KN_, (long)N_, nullptr, B_, N_, C_);

    // 6) softmax + sigmoid gate -> wprime = w * fw
    softmax_wprime<<<B_ * K_, 256, 0, stream>>>(simf, gbuf, bw, wprimeb);

    // 7) fE[b,c] = sum_{kk,n} wprime[b,kk,n] * Ev[kk,c,n]
    dim3 gf(C_ / 256, B_ / 64, 1);
    gemm_bf16<0, false><<<gf, 256, 0, stream>>>(wprimeb, KN_, EvT, KN_, 0,
                                                fEf, C_, 0, nullptr, B_, C_, KN_);

    // 8) final projection
    final_kernel<<<B_, 256, 0, stream>>>(vf, fEf, Wout, bout, out);
}
